// LSTMTagger_8022998909358
// MI455X (gfx1250) — matrix operand order repaired and checked by reading, not compiled, not run on hardware
//
#include <hip/hip_runtime.h>
#include <cstdint>
#include <cstddef>

// ---------------- problem constants ----------------
constexpr int T_LEN = 4096;
constexpr int EDIM  = 512;
constexpr int HDIM  = 1024;
constexpr int G4H   = 4096;   // 4*H
constexpr int NTAGS = 64;

// recurrence partitioning: 128 persistent workgroups, each owns 8 hidden units
constexpr int NWG_SCAN = 128;
constexpr int JPB      = 8;          // h-indices per block
constexpr int ROWS_PB  = 4 * JPB;    // 32 gate rows per block (i,f,g,o slices)

// ---------------- device vector types for WMMA ----------------
typedef __attribute__((ext_vector_type(8)))  __bf16 v8bf;
typedef __attribute__((ext_vector_type(16))) __bf16 v16bf;
typedef __attribute__((ext_vector_type(8)))  float  v8f;

__device__ __forceinline__ unsigned short f2bf(float f) {
    union { float f; unsigned u; } v; v.f = f;
    unsigned r = v.u + 0x7fffu + ((v.u >> 16) & 1u);   // round-to-nearest-even
    return (unsigned short)(r >> 16);
}

// =====================================================================
// Prep kernels: embedding gather -> bf16, weight conversion, bias merge,
// and barrier-counter reset (must happen every launch: graph replay safe).
// =====================================================================
__global__ void prep_gather_embed(const int* __restrict__ sentence,
                                  const float* __restrict__ emb,
                                  unsigned short* __restrict__ A_bf) {
    int idx = blockIdx.x * blockDim.x + threadIdx.x;     // over T*E
    if (idx < T_LEN * EDIM) {
        int t = idx >> 9;            // /512
        int e = idx & (EDIM - 1);
        A_bf[idx] = f2bf(emb[(size_t)sentence[t] * EDIM + e]);
    }
}

__global__ void prep_convert_bf16(const float* __restrict__ src,
                                  unsigned short* __restrict__ dst, int n) {
    int idx = blockIdx.x * blockDim.x + threadIdx.x;
    if (idx < n) dst[idx] = f2bf(src[idx]);
}

__global__ void prep_bias_ctr(const float* __restrict__ b_ih,
                              const float* __restrict__ b_hh,
                              float* __restrict__ bcomb,
                              unsigned* __restrict__ ctr) {
    int idx = blockIdx.x * blockDim.x + threadIdx.x;
    if (idx < G4H) bcomb[idx] = b_ih[idx] + b_hh[idx];
    if (idx == 0) *ctr = 0u;                              // reset step barrier
}

// =====================================================================
// GEMM 1: xp[T,4H] = x_bf16 @ W_ih_bf16^T + (b_ih+b_hh), fp32 accumulate.
// One wave computes a 16x16 tile via v_wmma_f32_16x16x32_bf16.
// Block = 8 waves -> 32(M) x 64(N) region. Grid = (4H/64, T/32).
// =====================================================================
__global__ __launch_bounds__(256)
void gemm_xproj(const unsigned short* __restrict__ Au16,
                const unsigned short* __restrict__ Wu16,
                const float* __restrict__ bias,
                float* __restrict__ xp) {
    const __bf16* A = reinterpret_cast<const __bf16*>(Au16);
    const __bf16* W = reinterpret_cast<const __bf16*>(Wu16);

    const int lane = threadIdx.x & 31;
    const int wave = threadIdx.x >> 5;
    const int kh   = lane >> 4;        // lane half (K sub-chunk select)
    const int sub  = lane & 15;        // M index (A) / N index (B)

    const int row0 = blockIdx.y * 32 + (wave >> 2) * 16;
    const int col0 = blockIdx.x * 64 + (wave & 3) * 16;

    // A (16x32 bf16): lane half 0 -> K 0..15 contiguous, half 1 -> K 16..31
    const __bf16* arow = A + (size_t)(row0 + sub) * EDIM + kh * 16;
    // B (32x16 bf16): lane half 0 -> K 0..15 contiguous, half 1 -> K 16..31
    const __bf16* brow = W + (size_t)(col0 + sub) * EDIM + kh * 16;

    v8f c = {};
    for (int k0 = 0; k0 < EDIM; k0 += 32) {
        __builtin_prefetch((const void*)(arow + k0 + 128), 0, 0);
        __builtin_prefetch((const void*)(brow + k0 + 128), 0, 0);
        v8bf al = *(const v8bf*)(arow + k0);
        v8bf ah = *(const v8bf*)(arow + k0 + 8);
        v16bf a, b;
        b = *(const v16bf*)(brow + k0);
#pragma unroll
        for (int i = 0; i < 8; ++i) { a[i] = al[i]; a[i + 8] = ah[i]; }
        c = __builtin_amdgcn_wmma_f32_16x16x32_bf16(
                false, a, false, b, (short)0, c, false, false);
    }

    const int col = col0 + sub;
    const float bb = bias[col];
#pragma unroll
    for (int r = 0; r < 8; ++r) {
        int m = row0 + r + kh * 8;                     // C/D layout
        xp[(size_t)m * G4H + col] = c[r] + bb;
    }
}

// =====================================================================
// Persistent LSTM recurrence. 128 blocks x 256 threads.
// Each block owns JPB=8 hidden units -> 32 rows of W_hh cached in LDS
// (128 KB of the 320 KB WGP LDS). Per step:
//   - cooperatively load h (4 KB) from L2 into LDS
//   - 32 rows x 1024 dot products (8 lanes/row, shfl-tree reduce)
//   - gate nonlinearities, update c (LDS-resident), publish h slice
//   - global monotonic-counter barrier across the 128 blocks
// =====================================================================
__global__ __launch_bounds__(256)
void lstm_scan(const float* __restrict__ xp,
               const float* __restrict__ Whh,
               float* __restrict__ h_buf,
               unsigned short* __restrict__ hs_bf,
               unsigned* __restrict__ ctr) {
    extern __shared__ float smem[];
    float* Ws    = smem;                        // [32][1024]
    float* h_sh  = Ws + ROWS_PB * HDIM;         // [1024]
    float* gates = h_sh + HDIM;                 // [32]
    float* cs    = gates + ROWS_PB;             // [8]

    const int tid = threadIdx.x;
    const int blk = blockIdx.x;

    // Stage this block's 32 rows of W_hh into LDS (float4 vectorized).
    for (int idx = tid; idx < (ROWS_PB * HDIM) / 4; idx += 256) {
        int flat = idx * 4;
        int r    = flat >> 10;                  // local row 0..31
        int cwi  = flat & (HDIM - 1);
        int grow = (r >> 3) * HDIM + blk * JPB + (r & 7);   // gate*H + owned j
        *(float4*)(Ws + flat) = *(const float4*)(Whh + (size_t)grow * HDIM + cwi);
    }
    if (tid < JPB) cs[tid] = 0.f;
    __syncthreads();

    volatile unsigned* vctr = ctr;
    const int r    = tid >> 3;                  // local gate row 0..31
    const int seg  = tid & 7;                   // 8 lanes per row
    const int grow = (r >> 3) * HDIM + blk * JPB + (r & 7);

    for (int t = 0; t < T_LEN; ++t) {
        float dot = 0.f;
        if (t > 0) {                            // h_{-1} == 0: skip matvec at t=0
            *(float4*)(h_sh + tid * 4) = *(const float4*)(h_buf + tid * 4);
            __syncthreads();
            const float* wr = Ws + r * HDIM + seg * 128;
            const float* hr = h_sh + seg * 128;
#pragma unroll 8
            for (int k = 0; k < 128; ++k) dot = __fmaf_rn(wr[k], hr[k], dot);
        }
        for (int o = 4; o; o >>= 1) dot += __shfl_down(dot, o, 8);
        if (seg == 0) gates[r] = xp[(size_t)t * G4H + grow] + dot;
        __syncthreads();

        if (tid < JPB) {
            float iv = gates[tid];
            float fv = gates[JPB + tid];
            float gv = gates[2 * JPB + tid];
            float ov = gates[3 * JPB + tid];
            iv = 1.f / (1.f + __expf(-iv));
            fv = 1.f / (1.f + __expf(-fv));
            gv = tanhf(gv);
            ov = 1.f / (1.f + __expf(-ov));
            float cc = fv * cs[tid] + iv * gv;
            cs[tid] = cc;
            float hh = ov * tanhf(cc);
            int jg = blk * JPB + tid;
            h_buf[jg] = hh;
            hs_bf[(size_t)t * HDIM + jg] = f2bf(hh);
        }
        __threadfence();
        __syncthreads();
        if (tid == 0) {
            atomicAdd(ctr, 1u);
            unsigned target = (unsigned)(t + 1) * NWG_SCAN;
            while (*vctr < target) { }
        }
        __syncthreads();
        __threadfence();
    }
}

// =====================================================================
// GEMM 2 + log_softmax: tags[T,64] = hs_bf16 @ W_out_bf16^T + b_out,
// then log_softmax over the 64 tags. Block = 8 waves -> 32 rows x 64 tags.
// =====================================================================
__global__ __launch_bounds__(256)
void gemm_tags_lsm(const unsigned short* __restrict__ HSu16,
                   const unsigned short* __restrict__ WOu16,
                   const float* __restrict__ b_out,
                   float* __restrict__ out) {
    const __bf16* HS = reinterpret_cast<const __bf16*>(HSu16);
    const __bf16* WO = reinterpret_cast<const __bf16*>(WOu16);

    __shared__ float tile[32][NTAGS + 1];

    const int lane = threadIdx.x & 31;
    const int wave = threadIdx.x >> 5;
    const int kh   = lane >> 4;
    const int sub  = lane & 15;

    const int row0 = blockIdx.x * 32 + (wave >> 2) * 16;
    const int col0 = (wave & 3) * 16;

    const __bf16* arow = HS + (size_t)(row0 + sub) * HDIM + kh * 16;
    const __bf16* brow = WO + (size_t)(col0 + sub) * HDIM + kh * 16;

    v8f c = {};
    for (int k0 = 0; k0 < HDIM; k0 += 32) {
        __builtin_prefetch((const void*)(arow + k0 + 128), 0, 0);
        v8bf al = *(const v8bf*)(arow + k0);
        v8bf ah = *(const v8bf*)(arow + k0 + 8);
        v16bf a, b;
        b = *(const v16bf*)(brow + k0);
#pragma unroll
        for (int i = 0; i < 8; ++i) { a[i] = al[i]; a[i + 8] = ah[i]; }
        c = __builtin_amdgcn_wmma_f32_16x16x32_bf16(
                false, a, false, b, (short)0, c, false, false);
    }

    const int col = col0 + sub;
    const float bb = b_out[col];
#pragma unroll
    for (int r = 0; r < 8; ++r)
        tile[(wave >> 2) * 16 + r + kh * 8][col] = c[r] + bb;
    __syncthreads();

    // log_softmax: 8 lanes per row (wave32-safe shfl width 8)
    const int rr  = threadIdx.x >> 3;   // 0..31
    const int seg = threadIdx.x & 7;
    float mx = -INFINITY;
#pragma unroll
    for (int i = 0; i < 8; ++i) mx = fmaxf(mx, tile[rr][seg * 8 + i]);
    for (int o = 1; o < 8; o <<= 1) mx = fmaxf(mx, __shfl_xor(mx, o, 8));
    float s = 0.f;
#pragma unroll
    for (int i = 0; i < 8; ++i) s += __expf(tile[rr][seg * 8 + i] - mx);
    for (int o = 1; o < 8; o <<= 1) s += __shfl_xor(s, o, 8);
    const float lse = mx + __logf(s);
    const int trow = blockIdx.x * 32 + rr;
#pragma unroll
    for (int i = 0; i < 8; ++i)
        out[(size_t)trow * NTAGS + seg * 8 + i] = tile[rr][seg * 8 + i] - lse;
}

// =====================================================================
// Host launcher
// =====================================================================
extern "C" void kernel_launch(void* const* d_in, const int* in_sizes, int n_in,
                              void* d_out, int out_size, void* d_ws, size_t ws_size,
                              hipStream_t stream) {
    const int*   sentence = (const int*)  d_in[0];
    const float* emb      = (const float*)d_in[1];
    const float* W_ih     = (const float*)d_in[2];
    const float* W_hh     = (const float*)d_in[3];
    const float* b_ih     = (const float*)d_in[4];
    const float* b_hh     = (const float*)d_in[5];
    const float* W_out    = (const float*)d_in[6];
    const float* b_out    = (const float*)d_in[7];
    float* out = (float*)d_out;

    // workspace carve (256-B aligned slabs)
    size_t off = 0;
    auto carve = [&](size_t bytes) -> void* {
        void* p = (char*)d_ws + off;
        off += (bytes + 255) & ~(size_t)255;
        return p;
    };
    unsigned short* A_bf    = (unsigned short*)carve((size_t)T_LEN * EDIM * 2);
    unsigned short* Wih_bf  = (unsigned short*)carve((size_t)G4H * EDIM * 2);
    unsigned short* Wout_bf = (unsigned short*)carve((size_t)NTAGS * HDIM * 2);
    unsigned short* hs_bf   = (unsigned short*)carve((size_t)T_LEN * HDIM * 2);
    float*          xp      = (float*)carve((size_t)T_LEN * G4H * 4);
    float*          bcomb   = (float*)carve((size_t)G4H * 4);
    float*          h_buf   = (float*)carve((size_t)HDIM * 4);
    unsigned*       ctr     = (unsigned*)carve(256);
    (void)ws_size; (void)in_sizes; (void)n_in; (void)out_size;

    // prep
    prep_gather_embed<<<(T_LEN * EDIM + 255) / 256, 256, 0, stream>>>(sentence, emb, A_bf);
    prep_convert_bf16<<<(G4H * EDIM + 255) / 256, 256, 0, stream>>>(W_ih, Wih_bf, G4H * EDIM);
    prep_convert_bf16<<<(NTAGS * HDIM + 255) / 256, 256, 0, stream>>>(W_out, Wout_bf, NTAGS * HDIM);
    prep_bias_ctr<<<(G4H + 255) / 256, 256, 0, stream>>>(b_ih, b_hh, bcomb, ctr);

    // input projection GEMM (WMMA bf16, fp32 accumulate)
    gemm_xproj<<<dim3(G4H / 64, T_LEN / 32), 256, 0, stream>>>(A_bf, Wih_bf, bcomb, xp);

    // persistent LSTM recurrence (W_hh resident in LDS across 128 WGPs)
    size_t smem = (size_t)(ROWS_PB * HDIM + HDIM + ROWS_PB + JPB) * sizeof(float);
    lstm_scan<<<NWG_SCAN, 256, smem, stream>>>(xp, W_hh, h_buf, hs_bf, ctr);

    // output projection + log_softmax (WMMA bf16)
    gemm_tags_lsm<<<T_LEN / 32, 256, 0, stream>>>(hs_bf, Wout_bf, b_out, out);
}
